// HQQDORA_26989574488653
// MI455X (gfx1250) — matrix operand order repaired and checked by reading, not compiled, not run on hardware
//
#include <hip/hip_runtime.h>
#include <hip/hip_bf16.h>

typedef __bf16 v16bf __attribute__((ext_vector_type(16)));
typedef __bf16 v8bf  __attribute__((ext_vector_type(8)));
typedef float  v8f   __attribute__((ext_vector_type(8)));

#define Mc 4096   // B*S
#define Nc 4096   // DOUT
#define Kc 4096   // DIN
#define Rr 32     // LoRA rank

// ------------------------------------------------------------------
// Kernel 1: scale[o] = magnitude[o] / || W[o,:] + lora_B[o,:] @ lora_A ||_2
// ------------------------------------------------------------------
#define S1_CH 256
__global__ void __launch_bounds__(256)
dora_scale_kernel(const float* __restrict__ W, const float* __restrict__ lA,
                  const float* __restrict__ lB, const float* __restrict__ mag,
                  float* __restrict__ scale) {
  __shared__ float sA[Rr][S1_CH];
  const int tid  = threadIdx.x;
  const int wave = tid >> 5, lane = tid & 31;
  const int o    = blockIdx.x * 8 + wave;

  float bl[Rr];
  #pragma unroll
  for (int r = 0; r < Rr; ++r) bl[r] = lB[(size_t)o * Rr + r];

  float acc = 0.f;
  for (int k0 = 0; k0 < Kc; k0 += S1_CH) {
    __syncthreads();
    for (int t = tid; t < Rr * S1_CH; t += 256) {
      const int r = t / S1_CH, c = t % S1_CH;
      sA[r][c] = lA[(size_t)r * Kc + k0 + c];
    }
    __syncthreads();
    for (int c = lane; c < S1_CH; c += 32) {
      float v = W[(size_t)o * Kc + k0 + c];
      #pragma unroll
      for (int r = 0; r < Rr; ++r) v = fmaf(bl[r], sA[r][c], v);
      acc = fmaf(v, v, acc);
    }
  }
  #pragma unroll
  for (int off = 16; off > 0; off >>= 1) acc += __shfl_xor(acc, off, 32);
  if (lane == 0) scale[o] = mag[o] / sqrtf(acc);
}

// ------------------------------------------------------------------
// Kernel 2: xa[m,r] = sum_i x[m,i] * lora_A[r,i]   ([M x 32])
// ------------------------------------------------------------------
#define S2_CH 128
__global__ void __launch_bounds__(256)
dora_xa_kernel(const float* __restrict__ x, const float* __restrict__ lA,
               float* __restrict__ xa) {
  __shared__ float sA[Rr][S2_CH];
  __shared__ float red[8][32][Rr + 1];
  const int tid  = threadIdx.x;
  const int wave = tid >> 5, lane = tid & 31;
  const int m    = blockIdx.x * 8 + wave;

  float acc[Rr];
  #pragma unroll
  for (int r = 0; r < Rr; ++r) acc[r] = 0.f;

  for (int k0 = 0; k0 < Kc; k0 += S2_CH) {
    __syncthreads();
    for (int t = tid; t < Rr * S2_CH; t += 256) {
      const int r = t / S2_CH, c = t % S2_CH;
      sA[r][c] = lA[(size_t)r * Kc + k0 + c];
    }
    __syncthreads();
    for (int c = lane; c < S2_CH; c += 32) {
      const float xv = x[(size_t)m * Kc + k0 + c];
      #pragma unroll
      for (int r = 0; r < Rr; ++r) acc[r] = fmaf(xv, sA[r][c], acc[r]);
    }
  }
  #pragma unroll
  for (int r = 0; r < Rr; ++r) red[wave][lane][r] = acc[r];
  __syncthreads();
  float s = 0.f;
  #pragma unroll
  for (int j = 0; j < 32; ++j) s += red[wave][j][lane];
  xa[(size_t)m * Rr + lane] = s;
}

// ------------------------------------------------------------------
// Kernel 3: out = (x @ W^T + xa @ lora_B^T) * scale[n]
// bf16x3 fp32-emulated WMMA GEMM, 128x128 tile / WG, 32x64 per wave.
// Ping-pong LDS double buffer, register-staged global loads,
// one barrier per K-step.
// ------------------------------------------------------------------
#define KS    32                  // K chunk (bf16 WMMA depth)
#define LDP   40                  // padded LDS row length (bf16 elements)
#define PLANE (128 * LDP)         // one bf16 plane (elements)
#define BUFE  (4 * PLANE)         // elements per ping-pong buffer (Ah,Al,Bh,Bl)
#define NKS   (Kc / KS + 1)       // 128 GEMM steps + 1 rank-32 LoRA tail

struct Stage { float4 a[4]; float4 b[4]; };

__device__ __forceinline__ void load_step(int s, int tid, int m0, int n0,
                                          const float* __restrict__ x,
                                          const float* __restrict__ W,
                                          const float* __restrict__ xa,
                                          const float* __restrict__ lB,
                                          Stage& st) {
  const int rlb = tid >> 3;           // row 0..31 (+p*32)
  const int c4  = (tid & 7) * 4;      // col 0,4,...,28
  if (s < Kc / KS) {
    const int k0 = s * KS;
    #pragma unroll
    for (int p = 0; p < 4; ++p) {
      const int rl = p * 32 + rlb;
      st.a[p] = *(const float4*)(x + (size_t)(m0 + rl) * Kc + k0 + c4);
      st.b[p] = *(const float4*)(W + (size_t)(n0 + rl) * Kc + k0 + c4);
    }
  } else {
    // LoRA tail: A-source = xa [M x 32], B-source = lora_B [N x 32]
    #pragma unroll
    for (int p = 0; p < 4; ++p) {
      const int rl = p * 32 + rlb;
      st.a[p] = *(const float4*)(xa + (size_t)(m0 + rl) * Rr + c4);
      st.b[p] = *(const float4*)(lB + (size_t)(n0 + rl) * Rr + c4);
    }
  }
}

__device__ __forceinline__ void cvt_store(const Stage& st, int tid, __bf16* base) {
  __bf16* Ah = base;
  __bf16* Al = base + PLANE;
  __bf16* Bh = base + 2 * PLANE;
  __bf16* Bl = base + 3 * PLANE;
  const int rlb = tid >> 3;
  const int c4  = (tid & 7) * 4;
  #pragma unroll
  for (int p = 0; p < 4; ++p) {
    const int idx = (p * 32 + rlb) * LDP + c4;
    const float fa[4] = {st.a[p].x, st.a[p].y, st.a[p].z, st.a[p].w};
    const float fb[4] = {st.b[p].x, st.b[p].y, st.b[p].z, st.b[p].w};
    #pragma unroll
    for (int j = 0; j < 4; ++j) {
      __bf16 h = (__bf16)fa[j];
      Ah[idx + j] = h;
      Al[idx + j] = (__bf16)(fa[j] - (float)h);
      h = (__bf16)fb[j];
      Bh[idx + j] = h;
      Bl[idx + j] = (__bf16)(fb[j] - (float)h);
    }
  }
}

__device__ __forceinline__ v16bf frag16(const __bf16* s, int row, int off0, int off1) {
  const v8bf a = *(const v8bf*)(s + row * LDP + off0);
  const v8bf b = *(const v8bf*)(s + row * LDP + off1);
  v16bf r;
  #pragma unroll
  for (int i = 0; i < 8; ++i) { r[i] = a[i]; r[i + 8] = b[i]; }
  return r;
}

__device__ __forceinline__ void compute_step(const __bf16* base, v8f (&acc)[2][4],
                                             int arow, int akoff, int bkoff,
                                             int wave_n, int lane) {
  const __bf16* Ahp = base;
  const __bf16* Alp = base + PLANE;
  const __bf16* Bhp = base + 2 * PLANE;
  const __bf16* Blp = base + 3 * PLANE;

  v16bf Ah[2], Al[2];
  #pragma unroll
  for (int mt = 0; mt < 2; ++mt) {
    Ah[mt] = frag16(Ahp, arow + mt * 16, akoff, 16 + akoff);
    Al[mt] = frag16(Alp, arow + mt * 16, akoff, 16 + akoff);
  }
  #pragma unroll
  for (int nt = 0; nt < 4; ++nt) {
    const int brow = wave_n * 64 + nt * 16 + (lane & 15);
    const v16bf Bh = frag16(Bhp, brow, bkoff, 16 + bkoff);
    const v16bf Bl = frag16(Blp, brow, bkoff, 16 + bkoff);
    // pass-major ordering: dependence distance 2 on each accumulator
    #pragma unroll
    for (int mt = 0; mt < 2; ++mt)
      acc[mt][nt] = __builtin_amdgcn_wmma_f32_16x16x32_bf16(
          false, Ah[mt], false, Bh, (short)0, acc[mt][nt], false, false);
    #pragma unroll
    for (int mt = 0; mt < 2; ++mt)
      acc[mt][nt] = __builtin_amdgcn_wmma_f32_16x16x32_bf16(
          false, Ah[mt], false, Bl, (short)0, acc[mt][nt], false, false);
    #pragma unroll
    for (int mt = 0; mt < 2; ++mt)
      acc[mt][nt] = __builtin_amdgcn_wmma_f32_16x16x32_bf16(
          false, Al[mt], false, Bh, (short)0, acc[mt][nt], false, false);
  }
}

__global__ void __launch_bounds__(256)
dora_gemm_kernel(const float* __restrict__ x, const float* __restrict__ W,
                 const float* __restrict__ xa, const float* __restrict__ lB,
                 const float* __restrict__ scale, float* __restrict__ out) {
  extern __shared__ __align__(16) __bf16 smem[];   // 2 * BUFE bf16 = 80 KB

  const int tid    = threadIdx.x;
  const int lane   = tid & 31, wave = tid >> 5;
  const int wave_m = wave & 3, wave_n = wave >> 2;     // 4 x 2 waves
  const int m0     = blockIdx.y * 128, n0 = blockIdx.x * 128;

  v8f acc[2][4];
  #pragma unroll
  for (int mt = 0; mt < 2; ++mt)
    #pragma unroll
    for (int nt = 0; nt < 4; ++nt)
      #pragma unroll
      for (int e = 0; e < 8; ++e) acc[mt][nt][e] = 0.f;

  // ISA bf16 A 16x32 layout: lane holds row (lane&15), K = (lane>>4)*8 + {0..7, 16..23}
  // ISA bf16 B 32x16 layout: lane holds col (lane&15), K = (lane>>4)*8 + {0..7, 16..23}
  const int arow  = wave_m * 32 + (lane & 15);
  const int akoff = (lane >> 4) * 8;
  const int bkoff = (lane >> 4) * 8;

  Stage st;
  load_step(0, tid, m0, n0, x, W, xa, lB, st);
  cvt_store(st, tid, smem);
  __syncthreads();

  for (int s = 0; s < NKS; ++s) {
    const __bf16* cur = smem + (s & 1) * BUFE;
    __bf16* nxt       = smem + ((s + 1) & 1) * BUFE;

    if (s + 1 < NKS) {
      load_step(s + 1, tid, m0, n0, x, W, xa, lB, st);   // overlaps WMMAs below
      if (s + 2 < Kc / KS) {                             // prefetch step s+2
        const int rl = tid >> 1;
        const int cc = (tid & 1) * 16;
        __builtin_prefetch(x + (size_t)(m0 + rl) * Kc + (s + 2) * KS + cc, 0, 1);
        __builtin_prefetch(W + (size_t)(n0 + rl) * Kc + (s + 2) * KS + cc, 0, 1);
      }
    }

    compute_step(cur, acc, arow, akoff, bkoff, wave_n, lane);

    if (s + 1 < NKS) cvt_store(st, tid, nxt);            // other buffer: no hazard
    __syncthreads();
  }

  // Epilogue: C/D 16x16 f32 layout: element e -> row (lane>>4)*8 + e, col lane&15
  #pragma unroll
  for (int nt = 0; nt < 4; ++nt) {
    const int n    = n0 + wave_n * 64 + nt * 16 + (lane & 15);
    const float sc = scale[n];
    #pragma unroll
    for (int mt = 0; mt < 2; ++mt) {
      const int mb = m0 + wave_m * 32 + mt * 16 + (lane >> 4) * 8;
      #pragma unroll
      for (int e = 0; e < 8; ++e)
        out[(size_t)(mb + e) * Nc + n] = acc[mt][nt][e] * sc;
    }
  }
}

// ------------------------------------------------------------------
extern "C" void kernel_launch(void* const* d_in, const int* in_sizes, int n_in,
                              void* d_out, int out_size, void* d_ws, size_t ws_size,
                              hipStream_t stream) {
  const float* x   = (const float*)d_in[0];
  const float* W   = (const float*)d_in[1];
  const float* lA  = (const float*)d_in[2];
  const float* lB  = (const float*)d_in[3];
  const float* mag = (const float*)d_in[4];
  float* out = (float*)d_out;

  float* xa    = (float*)d_ws;                 // M * R floats (512 KB)
  float* scale = xa + (size_t)Mc * Rr;         // N floats (16 KB)

  const int smem_bytes = 2 * BUFE * (int)sizeof(__bf16);   // 81920
  (void)hipFuncSetAttribute((const void*)dora_gemm_kernel,
                            hipFuncAttributeMaxDynamicSharedMemorySize, smem_bytes);

  dora_scale_kernel<<<Nc / 8, 256, 0, stream>>>(W, lA, lB, mag, scale);
  dora_xa_kernel   <<<Mc / 8, 256, 0, stream>>>(x, lA, xa);
  dora_gemm_kernel <<<dim3(Nc / 128, Mc / 128), 256, smem_bytes, stream>>>(
      x, W, xa, lB, scale, out);
}
